// MLSAttention_25812753449398
// MI455X (gfx1250) — matrix operand order repaired and checked by reading, not compiled, not run on hardware
//
#include <hip/hip_runtime.h>
#include <cmath>

typedef _Float16 v16h __attribute__((ext_vector_type(16)));
typedef _Float16 v8h  __attribute__((ext_vector_type(8)));
typedef float    v8f  __attribute__((ext_vector_type(8)));

#define SQ     4096
#define NH     16
#define HD     128
#define DEG    4
#define NT     1024            // selected keys per head = SQ/DEG
#define ROWSTR (NH*HD)         // 2048 floats per (s,b) row
#define MT     128             // query rows per block (8 waves x 16)
#define KC     32              // keys per LDS chunk
#define NCH    (NT/KC)         // 32 chunks
#define KROW   (HD+8)          // padded K row (halves): conflict-free ds_load_b128
#define VROW   (KC+8)          // padded V^T row (halves)

union PackHU { _Float16 h[2]; unsigned u; };
union B16U   { unsigned u[8]; v16h v; };

// ---- issue global loads for one 32-key chunk into registers (no LDS yet) ----
__device__ __forceinline__
void chunk_prefetch(const float* __restrict__ K, const float* __restrict__ V,
                    int t0, int wave, int lane, int hmod, int hbase,
                    float4 rk[4], float rv[16])
{
#pragma unroll
    for (int it = 0; it < 4; ++it) {
        int row = ((t0 + wave + 8*it) * DEG + hmod) * ROWSTR + hbase;
        rk[it] = *(const float4*)(K + row + lane * 4);        // 512B/wave, coalesced
#pragma unroll
        for (int jj = 0; jj < 4; ++jj)
            rv[it*4 + jj] = V[row + lane + 32*jj];            // 128B/wave, coalesced
    }
}

// ---- convert prefetched registers to f16 and store into one LDS buffer ----
__device__ __forceinline__
void chunk_store(_Float16* __restrict__ bK, _Float16* __restrict__ bVt,
                 int wave, int lane, const float4 rk[4], const float rv[16])
{
#pragma unroll
    for (int it = 0; it < 4; ++it) {
        _Float16* dk = bK + (wave + 8*it) * KROW + lane * 4;  // banks 2l..2l+1: conflict-free
        dk[0] = (_Float16)rk[it].x; dk[1] = (_Float16)rk[it].y;
        dk[2] = (_Float16)rk[it].z; dk[3] = (_Float16)rk[it].w;
#pragma unroll
        for (int jj = 0; jj < 4; ++jj)                        // rows lane+32jj: 2-way worst case
            bVt[(lane + 32*jj) * VROW + wave + 8*it] = (_Float16)rv[it*4 + jj];
    }
}

__global__ void __launch_bounds__(256)
mlsa_fwd_kernel(const float* __restrict__ Q, const float* __restrict__ K,
                const float* __restrict__ V, const unsigned char* __restrict__ MASK,
                float* __restrict__ OUT)
{
    __shared__ _Float16 sK [2][KC * KROW];   // K_sel chunk, row-major [key][d]
    __shared__ _Float16 sVt[2][HD * VROW];   // V_sel chunk, transposed [d][key]

    const int tid  = threadIdx.x;
    const int lane = tid & 31;
    const int wave = tid >> 5;
    const int q16  = lane & 15;           // query within wave tile / M row within A tile
    const int hi   = lane >> 4;           // half-wave (K-halves split)
    const int h    = blockIdx.y;
    const int hmod = h & (DEG - 1);
    const int hbase = h * HD;
    const int srow = blockIdx.x * MT + wave * 16 + q16;   // this lane's query row

    // ---- Q resident in registers as gemm1 B-fragments (Q^T columns), f32 -> f16
    const float* qrow = Q + (size_t)srow * ROWSTR + hbase;
    v16h bq[4];
#pragma unroll
    for (int j = 0; j < 4; ++j) {
        const float* p = qrow + 32 * j + hi * 8;
        float4 f0 = *(const float4*)(p + 0);
        float4 f1 = *(const float4*)(p + 4);
        float4 f2 = *(const float4*)(p + 16);
        float4 f3 = *(const float4*)(p + 20);
        v16h b;
        b[0]=(_Float16)f0.x;  b[1]=(_Float16)f0.y;  b[2]=(_Float16)f0.z;  b[3]=(_Float16)f0.w;
        b[4]=(_Float16)f1.x;  b[5]=(_Float16)f1.y;  b[6]=(_Float16)f1.z;  b[7]=(_Float16)f1.w;
        b[8]=(_Float16)f2.x;  b[9]=(_Float16)f2.y;  b[10]=(_Float16)f2.z; b[11]=(_Float16)f2.w;
        b[12]=(_Float16)f3.x; b[13]=(_Float16)f3.y; b[14]=(_Float16)f3.z; b[15]=(_Float16)f3.w;
        bq[j] = b;
    }

    v8f oacc[8] = {};          // O^T accumulators: 8 d-tiles x (16 d x 16 q)
    float dsum = 0.0f;         // per-lane partial softmax denominator

    // score -> log2-space: exp(s/sqrt(128)) == exp2(s * (log2e/sqrt(128)))
    const float SC2 = 0.08838834764831845f * 1.4426950408889634f;

    float4 rk[4]; float rv[16];
    chunk_prefetch(K, V, 0, wave, lane, hmod, hbase, rk, rv);

    for (int c = 0; c < NCH; ++c) {
        const int t0  = c * KC;
        const int buf = c & 1;

        // ---- drain prefetch into LDS buffer for this chunk
        chunk_store(sK[buf], sVt[buf], wave, lane, rk, rv);
        __syncthreads();

        // ---- issue next chunk's global loads; they fly during the WMMAs below
        if (c + 1 < NCH)
            chunk_prefetch(K, V, t0 + KC, wave, lane, hmod, hbase, rk, rv);

        const _Float16* bK  = sK[buf];
        const _Float16* bVt = sVt[buf];

        // ---- gemm1: S^T(16k x 16q) tiles = K_sel x Q^T over d=128 (4 x K=32)
        v8f s0 = {}, s1 = {};
#pragma unroll
        for (int j = 0; j < 4; ++j) {
            int lo = 32 * j + hi * 8;
            v8h r0a = *(const v8h*)&bK[q16 * KROW + lo];
            v8h r0b = *(const v8h*)&bK[q16 * KROW + lo + 16];
            v8h r1a = *(const v8h*)&bK[(16 + q16) * KROW + lo];
            v8h r1b = *(const v8h*)&bK[(16 + q16) * KROW + lo + 16];
            v16h a0, a1;
#pragma unroll
            for (int e = 0; e < 8; ++e) {
                a0[e] = r0a[e]; a0[8 + e] = r0b[e];
                a1[e] = r1a[e]; a1[8 + e] = r1b[e];
            }
            s0 = __builtin_amdgcn_wmma_f32_16x16x32_f16(false, a0, false, bq[j], (short)0, s0, false, false);
            s1 = __builtin_amdgcn_wmma_f32_16x16x32_f16(false, a1, false, bq[j], (short)0, s1, false, false);
        }

        // ---- scale+mask+exp in log2 space (v_mul + v_exp per element), pack to f16
        unsigned pk0[4], pk1[4];
        {
            const unsigned long long m0 =
                *(const unsigned long long*)(MASK + (size_t)srow * NT + t0 + hi * 8);
            const unsigned long long m1 =
                *(const unsigned long long*)(MASK + (size_t)srow * NT + t0 + 16 + hi * 8);
            PackHU u;
#pragma unroll
            for (int j = 0; j < 4; ++j) {
                float t, e0, e1;
                t = s0[2*j]   * SC2; if ((m0 >> (16*j))     & 0xFFull) t = -16384.0f;
                e0 = __builtin_amdgcn_exp2f(t);
                t = s0[2*j+1] * SC2; if ((m0 >> (16*j + 8)) & 0xFFull) t = -16384.0f;
                e1 = __builtin_amdgcn_exp2f(t);
                dsum += e0 + e1;
                u.h[0] = (_Float16)e0; u.h[1] = (_Float16)e1; pk0[j] = u.u;

                t = s1[2*j]   * SC2; if ((m1 >> (16*j))     & 0xFFull) t = -16384.0f;
                e0 = __builtin_amdgcn_exp2f(t);
                t = s1[2*j+1] * SC2; if ((m1 >> (16*j + 8)) & 0xFFull) t = -16384.0f;
                e1 = __builtin_amdgcn_exp2f(t);
                dsum += e0 + e1;
                u.h[0] = (_Float16)e0; u.h[1] = (_Float16)e1; pk1[j] = u.u;
            }
        }

        // ---- half-wave exchange: build P^T B-fragment (lane<16: keys 0..15, lane>=16: 16..31)
        B16U b2;
#pragma unroll
        for (int j = 0; j < 4; ++j) {
            unsigned pp0 = (unsigned)__shfl_xor((int)pk0[j], 16, 32);
            unsigned pp1 = (unsigned)__shfl_xor((int)pk1[j], 16, 32);
            b2.u[j]     = hi ? pp1    : pk0[j];
            b2.u[4 + j] = hi ? pk1[j] : pp0;
        }

        // ---- gemm2: O^T += V^T x P^T  (8 d-tiles, K=32 keys)
#pragma unroll
        for (int dt = 0; dt < 8; ++dt) {
            int dd = dt * 16 + q16;
            v8h va = *(const v8h*)&bVt[dd * VROW + hi * 16];
            v8h vb = *(const v8h*)&bVt[dd * VROW + hi * 16 + 8];
            v16h a;
#pragma unroll
            for (int e = 0; e < 8; ++e) { a[e] = va[e]; a[8 + e] = vb[e]; }
            oacc[dt] = __builtin_amdgcn_wmma_f32_16x16x32_f16(false, a, false, b2.v, (short)0, oacc[dt], false, false);
        }
    }

    // ---- epilogue: full denominator via half-wave reduce, normalize, coalesced store
    float dtot = dsum + __shfl_xor(dsum, 16, 32);
    float inv  = 1.0f / dtot;
    float* orow = OUT + (size_t)srow * ROWSTR + hbase + hi * 8;
#pragma unroll
    for (int dt = 0; dt < 8; ++dt) {
        float4 lo4 = make_float4(oacc[dt][0]*inv, oacc[dt][1]*inv, oacc[dt][2]*inv, oacc[dt][3]*inv);
        float4 hi4 = make_float4(oacc[dt][4]*inv, oacc[dt][5]*inv, oacc[dt][6]*inv, oacc[dt][7]*inv);
        *(float4*)(orow + dt * 16 + 0) = lo4;
        *(float4*)(orow + dt * 16 + 4) = hi4;
    }
}

extern "C" void kernel_launch(void* const* d_in, const int* in_sizes, int n_in,
                              void* d_out, int out_size, void* d_ws, size_t ws_size,
                              hipStream_t stream) {
    (void)in_sizes; (void)n_in; (void)d_ws; (void)ws_size; (void)out_size;
    const float* q = (const float*)d_in[0];
    const float* k = (const float*)d_in[1];
    const float* v = (const float*)d_in[2];
    const unsigned char* m = (const unsigned char*)d_in[3];
    float* out = (float*)d_out;

    dim3 grid(SQ / MT, NH);     // 32 query tiles x 16 heads
    dim3 block(256);            // 8 wave32s
    hipLaunchKernelGGL(mlsa_fwd_kernel, grid, block, 0, stream, q, k, v, m, out);
}
